// RNNModule_3238405341716
// MI455X (gfx1250) — matrix operand order repaired and checked by reading, not compiled, not run on hardware
//
#include <hip/hip_runtime.h>
#include <hip/hip_bf16.h>

// Problem dims (fixed by the reference)
#define B_ 128
#define T_ 1024
#define I_ 256
#define H_ 512

typedef __attribute__((ext_vector_type(16))) __bf16        v16bf;
typedef __attribute__((ext_vector_type(8)))  float         v8f;
typedef __attribute__((ext_vector_type(4)))  unsigned int  v4u;

union Frag16 { v16bf bf; v4u q[2]; };

// float -> bf16 with round-to-nearest-even
__device__ __forceinline__ unsigned short f2bf(float f) {
  unsigned int u = __builtin_bit_cast(unsigned int, f);
  u += 0x7FFFu + ((u >> 16) & 1u);
  return (unsigned short)(u >> 16);
}

__device__ __forceinline__ float sigmoidf(float v) {
  return 1.0f / (1.0f + __expf(-v));
}

// A-fragment (16x32 bf16, MxK) from an LDS tile, per ISA 7.12.2 layout:
// lane l: row m = l&15; chunk0 K = (l>>4)*8 .. +7 (VGPR0-3), chunk1 = +16 (VGPR4-7)
__device__ __forceinline__ v16bf load_a_frag(const unsigned short* As, int ldk,
                                             int kblk, int lane) {
  int m  = lane & 15;
  int hi = lane >> 4;
  const unsigned short* p = As + m * ldk + kblk + hi * 8;
  Frag16 f;
  f.q[0] = *reinterpret_cast<const v4u*>(p);        // K = base .. base+7
  f.q[1] = *reinterpret_cast<const v4u*>(p + 16);   // K = base+16 .. base+23
  return f.bf;
}

// B-fragment (32x16 bf16, KxN) straight from row-major bf16 weights W[N][K]:
// B[k][n] = W[n][k]; lane l: col n = l&15, K = kblk + (l>>4)*8 .. +7, then +16
__device__ __forceinline__ v16bf load_b_frag(const unsigned short* __restrict__ W,
                                             int row_base, int K, int kblk, int lane) {
  int n  = lane & 15;
  int hi = lane >> 4;
  const unsigned short* p = W + (size_t)(row_base + n) * K + kblk + hi * 8;
  Frag16 f;
  f.q[0] = *reinterpret_cast<const v4u*>(p);
  f.q[1] = *reinterpret_cast<const v4u*>(p + 16);
  return f.bf;
}

// Elementwise fp32 -> bf16 weight conversion (one-time, tiny)
__global__ void cvt_f32_bf16(const float* __restrict__ src,
                             unsigned short* __restrict__ dst, int n) {
  int i = blockIdx.x * blockDim.x + threadIdx.x;
  if (i < n) dst[i] = f2bf(src[i]);
}

// Phase 1: igates = x @ Wi^T + bi, written fp32 into out[B*T, H] (in place;
// the scan overwrites each time-slice with h_t).
// Block = 256 thr = 8 waves; block tile = 16(M) x 128(N); wave tile = 16x16.
__global__ void __launch_bounds__(256)
igates_kernel(const float* __restrict__ x,              // [B*T, I]
              const unsigned short* __restrict__ Wi_bf, // [H, I] bf16
              const float* __restrict__ bi,             // [H]
              float* __restrict__ out)                  // [B*T, H]
{
  __shared__ __align__(16) unsigned short As[16 * I_];  // 8 KB bf16 A tile
  const int NBN  = H_ / 128;                            // 4 N-blocks
  int mtile = (blockIdx.x / NBN) * 16;
  int ntile = (blockIdx.x % NBN) * 128 + (threadIdx.x >> 5) * 16;
  int lane  = threadIdx.x & 31;

  // Stage 16x256 fp32 -> bf16 into LDS (coalesced: row i, col = tid)
  {
    const float* xp = x + (size_t)mtile * I_;
#pragma unroll
    for (int i = 0; i < 16; ++i)
      As[i * I_ + threadIdx.x] = f2bf(xp[(size_t)i * I_ + threadIdx.x]);
  }
  __syncthreads();

  v8f acc = {};
#pragma unroll
  for (int kb = 0; kb < I_ / 32; ++kb) {                // 8 bf16 WMMAs
    v16bf a = load_a_frag(As, I_, kb * 32, lane);
    v16bf b = load_b_frag(Wi_bf, ntile, I_, kb * 32, lane);
    acc = __builtin_amdgcn_wmma_f32_16x16x32_bf16(false, a, false, b,
                                                  (short)0, acc, false, false);
  }

  // C layout: lane covers column ntile+(lane&15); rows r + (lane>>4)*8
  int n  = lane & 15;
  int hi = lane >> 4;
  int ng = ntile + n;
  float bias = bi[ng];
#pragma unroll
  for (int r = 0; r < 8; ++r) {
    int mg = mtile + r + hi * 8;
    out[(size_t)mg * H_ + ng] = acc[r] + bias;
  }
}

// Phase 2: the entire T=1024 scan in ONE persistent kernel.
// Each block owns 16 batch rows and the FULL H=512 columns, so the
// recurrence h_t[b,:] = sigmoid(ig_t[b,:] + h_{t-1}[b,:]@Wh^T + bh) is
// block-local: no device-wide sync, one __syncthreads per step.
// h lives in LDS as bf16, double-buffered (2 x 16 KB). Wh (bf16, 512 KB)
// streams from L2 every step. 512 thr = 16 waves; wave owns 32 columns
// (two 16x16 tiles sharing each A fragment) -> 32 WMMAs/wave/step.
__global__ void __launch_bounds__(512)
rnn_scan_persistent(const float* __restrict__ hidden,        // [B, H]
                    const unsigned short* __restrict__ Wh_bf,// [H, H] bf16
                    const float* __restrict__ bh,            // [H]
                    float* __restrict__ out,                 // [B, T, H]
                    float* __restrict__ hlast)               // [B, H]
{
  __shared__ __align__(16) unsigned short Hs[2][16 * H_];    // 32 KB
  int mtile = blockIdx.x * 16;
  int wave  = threadIdx.x >> 5;
  int lane  = threadIdx.x & 31;
  int nt0   = wave * 32;          // two 16-col tiles per wave
  int nt1   = nt0 + 16;

  // Stage h_0 (the provided initial hidden state) as bf16 into buffer 0
  for (int i = threadIdx.x; i < 16 * H_; i += 512) {
    int row = i >> 9, col = i & (H_ - 1);
    Hs[0][i] = f2bf(hidden[(size_t)(mtile + row) * H_ + col]);
  }
  __syncthreads();

  int n   = lane & 15;
  int hi  = lane >> 4;
  int ng0 = nt0 + n, ng1 = nt1 + n;
  float bias0 = bh[ng0];
  float bias1 = bh[ng1];

  int cur = 0;
  for (int t = 0; t < T_; ++t) {
    int nxt = cur ^ 1;
    v8f acc0 = {}, acc1 = {};
#pragma unroll
    for (int kb = 0; kb < H_ / 32; ++kb) {               // 2 x 16 bf16 WMMAs
      v16bf a  = load_a_frag(Hs[cur], H_, kb * 32, lane);
      v16bf b0 = load_b_frag(Wh_bf, nt0, H_, kb * 32, lane);
      v16bf b1 = load_b_frag(Wh_bf, nt1, H_, kb * 32, lane);
      acc0 = __builtin_amdgcn_wmma_f32_16x16x32_bf16(false, a, false, b0,
                                                     (short)0, acc0, false, false);
      acc1 = __builtin_amdgcn_wmma_f32_16x16x32_bf16(false, a, false, b1,
                                                     (short)0, acc1, false, false);
    }

    // Epilogue: sigmoid(acc + ig + bias); write global h_t, stage bf16 h_t
    // into the next LDS buffer, prefetch next step's igates.
#pragma unroll
    for (int r = 0; r < 8; ++r) {
      int lrow = r + hi * 8;                             // row within tile
      int b    = mtile + lrow;                           // batch index
      size_t base = ((size_t)b * T_ + t) * H_;
      float v0 = sigmoidf(acc0[r] + out[base + ng0] + bias0);
      float v1 = sigmoidf(acc1[r] + out[base + ng1] + bias1);
      out[base + ng0] = v0;
      out[base + ng1] = v1;
      Hs[nxt][lrow * H_ + ng0] = f2bf(v0);
      Hs[nxt][lrow * H_ + ng1] = f2bf(v1);
      if (t == T_ - 1) {
        hlast[(size_t)b * H_ + ng0] = v0;
        hlast[(size_t)b * H_ + ng1] = v1;
      } else {
        __builtin_prefetch(&out[base + H_ + ng0], 0, 1); // next-step igates
        __builtin_prefetch(&out[base + H_ + ng1], 0, 1);
      }
    }
    __syncthreads();
    cur = nxt;
  }
}

extern "C" void kernel_launch(void* const* d_in, const int* in_sizes, int n_in,
                              void* d_out, int out_size, void* d_ws, size_t ws_size,
                              hipStream_t stream) {
  (void)in_sizes; (void)n_in; (void)out_size; (void)ws_size;

  const float* x      = (const float*)d_in[0];  // [B,T,I]
  const float* hidden = (const float*)d_in[1];  // [B,H]
  const float* Wi     = (const float*)d_in[2];  // [H,I]
  const float* bi     = (const float*)d_in[3];  // [H]
  const float* Wh     = (const float*)d_in[4];  // [H,H]
  const float* bh     = (const float*)d_in[5];  // [H]

  float* out   = (float*)d_out;                       // [B,T,H] then [B,H]
  float* hlast = out + (size_t)B_ * T_ * H_;

  // Workspace: bf16 copies of Wh (512 KB) and Wi (256 KB)
  unsigned short* Wh_bf = (unsigned short*)d_ws;
  unsigned short* Wi_bf = Wh_bf + (size_t)H_ * H_;

  cvt_f32_bf16<<<(H_ * H_ + 255) / 256, 256, 0, stream>>>(Wh, Wh_bf, H_ * H_);
  cvt_f32_bf16<<<(H_ * I_ + 255) / 256, 256, 0, stream>>>(Wi, Wi_bf, H_ * I_);

  // Phase 1: igates into d_out (in place), device-wide parallel GEMM
  int gridI = (B_ * T_ / 16) * (H_ / 128);   // 8192 * 4 = 32768 blocks
  igates_kernel<<<gridI, 256, 0, stream>>>(x, Wi_bf, bi, out);

  // Phase 2: whole scan in one persistent launch (8 blocks, batch-parallel)
  rnn_scan_persistent<<<B_ / 16, 512, 0, stream>>>(hidden, Wh_bf, bh, out, hlast);
}
